// GCNConv_15195594293515
// MI455X (gfx1250) — matrix operand order repaired and checked by reading, not compiled, not run on hardware
//
#include <hip/hip_runtime.h>

typedef __attribute__((ext_vector_type(16))) _Float16 v16h;
typedef __attribute__((ext_vector_type(8)))  float    v8f;

#define F_DIM 128   // F_IN == F_OUT == 128

// ---------------------------------------------------------------------------
// Kernel 1: pack W (f32 row-major 128x128) into f16 WMMA-B fragment layout.
// Fragment (j,s): j = N-tile (0..7, 16 cols), s = K-step (0..3, 32 K values).
// Per ISA B-matrix striping (16-bit, 32x16): lane l -> col n = j*16 + (l&15);
// lanes 0-15 hold K = s*32 + 0..7,16..23, lanes 16-31 hold K = s*32 + 8..15,24..31,
// element e (0..15) -> K = s*32 + (e>>3)*16 + (l>>4)*8 + (e&7).
// Stored contiguously: Wp[((j*4+s)*32 + l)*16 + e]  (32 B per lane, coalesced).
// ---------------------------------------------------------------------------
__global__ __launch_bounds__(256) void pack_w_kernel(const float* __restrict__ W,
                                                     _Float16* __restrict__ Wp) {
    int idx  = blockIdx.x * 256 + threadIdx.x;   // 0..16383
    if (idx >= 32 * 32 * 16) return;
    int e    = idx & 15;
    int lane = (idx >> 4) & 31;
    int frag = idx >> 9;                         // 0..31 == j*4 + s
    int s    = frag & 3;
    int j    = frag >> 2;
    int n    = j * 16 + (lane & 15);
    int K    = s * 32 + (e >> 3) * 16 + (lane >> 4) * 8 + (e & 7);
    Wp[idx]  = (_Float16)W[K * F_DIM + n];
}

// ---------------------------------------------------------------------------
// Kernel 2: support = X @ W via v_wmma_f32_16x16x32_f16.
// Block = 256 threads = 8 waves; block computes a 128x128 output tile.
// Each wave owns a 16x128 strip: 8 accumulator tiles (8 x v8f = 64 VGPRs).
// A fragment per ISA 16-bit A layout: lane half 'hi' -> row = l&15,
//   elements 0..7  -> K = kb..kb+7      (kb = s*32 + hi*8)
//   elements 8..15 -> K = kb+16..kb+23
// Loaded straight from global X as two float4 pairs, converted f32->f16.
// ---------------------------------------------------------------------------
__global__ __launch_bounds__(256) void gemm_xw_kernel(const float* __restrict__ X,
                                                      const _Float16* __restrict__ Wp,
                                                      float* __restrict__ S,
                                                      int N) {
    const int wave = threadIdx.x >> 5;   // 0..7
    const int lane = threadIdx.x & 31;
    const int half = lane & 15;
    const int hi   = lane >> 4;          // 0 or 1

    const int rowBase = blockIdx.x * 128 + wave * 16;   // first M row of strip
    int row = rowBase + half;                            // A row this lane feeds
    int rowClamped = row < N ? row : (N - 1);            // keep loads in-bounds
    const float* xrow = X + (size_t)rowClamped * F_DIM;

    v8f acc[8];
#pragma unroll
    for (int j = 0; j < 8; ++j) acc[j] = (v8f){0.f,0.f,0.f,0.f,0.f,0.f,0.f,0.f};

#pragma unroll
    for (int s = 0; s < 4; ++s) {
        const int kb = s * 32 + hi * 8;
        float4 a0 = *(const float4*)(xrow + kb);
        float4 a1 = *(const float4*)(xrow + kb + 4);
        float4 a2 = *(const float4*)(xrow + kb + 16);
        float4 a3 = *(const float4*)(xrow + kb + 20);
        float af[16];
        *(float4*)(af + 0)  = a0;
        *(float4*)(af + 4)  = a1;
        *(float4*)(af + 8)  = a2;
        *(float4*)(af + 12) = a3;
        v16h a;
#pragma unroll
        for (int t = 0; t < 16; ++t) a[t] = (_Float16)af[t];

#pragma unroll
        for (int j = 0; j < 8; ++j) {
            const v16h b = *(const v16h*)(Wp + ((size_t)(j * 4 + s) * 32 + lane) * 16);
            // 8 args: (neg_a, A, neg_b, B, c_mod, C, reuse_a, reuse_b)
            acc[j] = __builtin_amdgcn_wmma_f32_16x16x32_f16(
                false, a, false, b, (short)0, acc[j], false, false);
        }
    }

    // C/D layout: VGPR r, lanes 0-15 -> M = r, lanes 16-31 -> M = 8 + r; N = lane&15.
    const int mBase = rowBase + hi * 8;
#pragma unroll
    for (int j = 0; j < 8; ++j) {
        const int col = j * 16 + half;
#pragma unroll
        for (int r = 0; r < 8; ++r) {
            const int m = mBase + r;
            if (m < N) S[(size_t)m * F_DIM + col] = acc[j][r];
        }
    }
}

// ---------------------------------------------------------------------------
// Kernel 3: zero the output (harness poisons d_out with 0xAA before timing).
// ---------------------------------------------------------------------------
__global__ __launch_bounds__(256) void zero_kernel(float4* __restrict__ out, int n4) {
    int stride = gridDim.x * blockDim.x;
    for (int i = blockIdx.x * blockDim.x + threadIdx.x; i < n4; i += stride)
        out[i] = (float4){0.f, 0.f, 0.f, 0.f};
}

// ---------------------------------------------------------------------------
// Kernel 4: SpMM scatter-add. One wave per edge; lane handles 4 features.
// support rows (51.2 MB) are L2-resident; gather is a coalesced 512 B/edge
// global_load_b128. Scatter is forced to the non-returning hardware atomic
// (global_atomic_add_f32, STOREcnt-tracked, RMW in the L2 atomic units) via
// inline CDNA5 asm — avoids any CAS-loop fallback and any TH_ATOMIC_RETURN
// traffic. Prefetch the next edge's support row to hide random L2 latency.
// ---------------------------------------------------------------------------
__global__ __launch_bounds__(256) void spmm_kernel(const float* __restrict__ S,
                                                   const float* __restrict__ vals,
                                                   const int* __restrict__ rows,
                                                   const int* __restrict__ cols,
                                                   float* __restrict__ out,
                                                   int E) {
    const int lane = threadIdx.x & 31;
    const int wid  = blockIdx.x * (blockDim.x >> 5) + (threadIdx.x >> 5);
    const int nw   = gridDim.x * (blockDim.x >> 5);

    for (int e = wid; e < E; e += nw) {
        if (e + nw < E) {
            int cn = cols[e + nw];
            __builtin_prefetch(S + (size_t)cn * F_DIM + lane * 4, 0, 0);
        }
        const int   r = rows[e];
        const int   c = cols[e];
        const float v = vals[e];
        const float4 sv = *(const float4*)(S + (size_t)c * F_DIM + lane * 4);
        float* op = out + (size_t)r * F_DIM + lane * 4;
        const float m0 = v * sv.x;
        const float m1 = v * sv.y;
        const float m2 = v * sv.z;
        const float m3 = v * sv.w;
        asm volatile(
            "global_atomic_add_f32 %0, %1, off scope:SCOPE_DEV\n\t"
            "global_atomic_add_f32 %0, %2, off offset:4 scope:SCOPE_DEV\n\t"
            "global_atomic_add_f32 %0, %3, off offset:8 scope:SCOPE_DEV\n\t"
            "global_atomic_add_f32 %0, %4, off offset:12 scope:SCOPE_DEV"
            :
            : "v"(op), "v"(m0), "v"(m1), "v"(m2), "v"(m3)
            : "memory");
    }
}

// ---------------------------------------------------------------------------
extern "C" void kernel_launch(void* const* d_in, const int* in_sizes, int n_in,
                              void* d_out, int out_size, void* d_ws, size_t ws_size,
                              hipStream_t stream) {
    const float* x  = (const float*)d_in[0];   // [N, 128] f32
    const float* W  = (const float*)d_in[1];   // [128, 128] f32
    const float* ev = (const float*)d_in[2];   // [E] f32
    const int*   er = (const int*)d_in[3];     // [E] i32
    const int*   ec = (const int*)d_in[4];     // [E] i32
    float* out = (float*)d_out;                // [N, 128] f32

    const int N = in_sizes[0] / F_DIM;
    const int E = in_sizes[2];

    // Workspace carve-out: [0, 32KB) packed f16 W; then f32 support [N,128].
    _Float16* Wp = (_Float16*)d_ws;
    float*    S  = (float*)((char*)d_ws + 32768);

    pack_w_kernel<<<64, 256, 0, stream>>>(W, Wp);

    const int gemmBlocks = (N + 127) / 128;
    gemm_xw_kernel<<<gemmBlocks, 256, 0, stream>>>(x, Wp, S, N);

    zero_kernel<<<2048, 256, 0, stream>>>((float4*)out, out_size / 4);

    spmm_kernel<<<2048, 256, 0, stream>>>(S, ev, er, ec, out, E);
}
